// SharedClassName_65481071400271
// MI455X (gfx1250) — compile-verified, matrix operand order checked by reading, not run on hardware
//
#include <hip/hip_runtime.h>
#include <math.h>

typedef __attribute__((ext_vector_type(16))) _Float16 v16h;
typedef __attribute__((ext_vector_type(8)))  float    v8f;

#define NQ     8
#define EMBED  64
#define HIDDEN 8
#define TAGSET 16
#define B_     128
#define T_     64
#define G32    32        // 4 gates * 8 qubit-outputs
#define XP_FLOATS (T_ * G32 * B_)        // 262144 floats = 1 MB

// ---------------------------------------------------------------------------
// Prep: pack the x-part of the 4 gate weight matrices (columns 0..63 of each
// (8,72) W) into f16 Wx[32][64], fold bias+theta into biasth[32], and pack the
// h-part into Wh[32][8].
// ---------------------------------------------------------------------------
__global__ void qlstm_prep(const float* __restrict__ Wf, const float* __restrict__ bf, const float* __restrict__ thf,
                           const float* __restrict__ Wi, const float* __restrict__ bi, const float* __restrict__ thi,
                           const float* __restrict__ Wu, const float* __restrict__ bu, const float* __restrict__ thu,
                           const float* __restrict__ Wo, const float* __restrict__ bo, const float* __restrict__ tho,
                           _Float16* __restrict__ Wx, float* __restrict__ biasth, float* __restrict__ Wh) {
    for (int g = threadIdx.x; g < G32; g += blockDim.x) {
        int gate = g >> 3, w = g & 7;
        const float* W  = gate == 0 ? Wf  : gate == 1 ? Wi  : gate == 2 ? Wu  : Wo;
        const float* bb = gate == 0 ? bf  : gate == 1 ? bi  : gate == 2 ? bu  : bo;
        const float* tt = gate == 0 ? thf : gate == 1 ? thi : gate == 2 ? thu : tho;
        for (int e = 0; e < EMBED; ++e)  Wx[g * EMBED + e]  = (_Float16)W[w * 72 + e];
        for (int j = 0; j < HIDDEN; ++j) Wh[g * HIDDEN + j] = W[w * 72 + EMBED + j];
        biasth[g] = bb[w] + tt[w];   // cos(z + th): fold bias and theta together
    }
}

// ---------------------------------------------------------------------------
// Time-parallel projection: Xp[t][g][b] = emb[sentence[b,t]] . Wx[g] + biasth[g]
// M = 8192 (t,b) rows, N = 32 gate outputs, K = 64 -> WMMA f16, K split in 2.
// One 16(row) x 32(col) output tile per wave; 4 waves/block, 128 blocks.
//
// Fragment packing per CDNA5 ISA 7.12.2 (16-bit A 16x32):
//   lane<16 : row M=lane,    halves 0..7 = K 0..7,  halves 8..15 = K 16..23
//   lane>=16: row M=lane-16, halves 0..7 = K 8..15, halves 8..15 = K 24..31
// B (32x16) mirrored with lane = N column; D: lane = N, slot r = M (+8 for hi lanes).
// ---------------------------------------------------------------------------
__global__ void qlstm_xproj(const int* __restrict__ sentence, const float* __restrict__ emb,
                            const _Float16* __restrict__ Wx, const float* __restrict__ biasth,
                            float* __restrict__ Xp) {
    int lane = threadIdx.x & 31;
    int tile = blockIdx.x * 4 + (threadIdx.x >> 5);   // 512 tiles total
    int t  = tile >> 3;
    int b0 = (tile & 7) << 4;
    int m   = lane & 15;      // row within tile (A) / column within tile (B,D)
    int sel = lane >> 4;      // which K half-group this lane owns

    int b = b0 + m;
    int idx = sentence[b * T_ + t];
    const float* erow = emb + (long long)idx * EMBED;

    v16h a[2];
    #pragma unroll
    for (int ks = 0; ks < 2; ++ks) {
        int e0 = ks * 32 + sel * 8;
        #pragma unroll
        for (int i = 0; i < 8; ++i) {
            a[ks][i]     = (_Float16)erow[e0 + i];
            a[ks][8 + i] = (_Float16)erow[e0 + 16 + i];
        }
    }

    #pragma unroll
    for (int nt = 0; nt < 2; ++nt) {
        int g = nt * 16 + m;
        const _Float16* wrow = Wx + g * EMBED;
        v8f acc = {};
        #pragma unroll
        for (int ks = 0; ks < 2; ++ks) {
            v16h bf;
            int e0 = ks * 32 + sel * 8;
            #pragma unroll
            for (int i = 0; i < 8; ++i) {
                bf[i]     = wrow[e0 + i];
                bf[8 + i] = wrow[e0 + 16 + i];
            }
            acc = __builtin_amdgcn_wmma_f32_16x16x32_f16(
                      false, a[ks], false, bf, (short)0, acc, false, false);
        }
        float bt = biasth[g];
        // lane holds column g; slots r=0..7 are rows b0+sel*8 .. b0+sel*8+7
        float* op = Xp + t * (G32 * B_) + g * B_ + b0 + sel * 8;
        #pragma unroll
        for (int r = 0; r < 8; ++r) op[r] = acc[r] + bt;
    }
}

// ---------------------------------------------------------------------------
// Sequential recurrence: one lane per batch element, 4 blocks x 32 lanes.
// Quantum layer collapsed analytically to cos-prefix-products:
//   C_j = cos(phi_j);  out[w>=1] = C_0*...*C_w ;  out[0] = C_1*...*C_7
// Fused tag projection + log_softmax -> d_out directly.
// ---------------------------------------------------------------------------
__global__ void qlstm_recur(const float* __restrict__ Xp, const float* __restrict__ Wh,
                            const float* __restrict__ Wtag, const float* __restrict__ btag,
                            float* __restrict__ out) {
    __shared__ float sWh[G32 * HIDDEN];
    __shared__ float sWt[TAGSET * HIDDEN];
    __shared__ float sbt[TAGSET];
    int lane = threadIdx.x;
    for (int i = lane; i < G32 * HIDDEN; i += 32)    sWh[i] = Wh[i];
    for (int i = lane; i < TAGSET * HIDDEN; i += 32) sWt[i] = Wtag[i];
    if (lane < TAGSET) sbt[lane] = btag[lane];
    __syncthreads();

    int b = blockIdx.x * 32 + lane;
    float h[HIDDEN], c[HIDDEN];
    #pragma unroll
    for (int j = 0; j < HIDDEN; ++j) { h[j] = 0.f; c[j] = 0.f; }

    for (int t = 0; t < T_; ++t) {
        const float* xp = Xp + t * (G32 * B_) + b;
        float C[G32];
        #pragma unroll
        for (int g = 0; g < G32; ++g) {
            float z = xp[g * B_];                       // coalesced across lanes
            #pragma unroll
            for (int j = 0; j < HIDDEN; ++j) z = fmaf(h[j], sWh[g * HIDDEN + j], z);
            C[g] = __cosf(z);
        }
        float q[4][NQ];
        #pragma unroll
        for (int gate = 0; gate < 4; ++gate) {
            const float* Cg = C + gate * 8;
            float pre = Cg[0];
            #pragma unroll
            for (int w = 1; w < 8; ++w) { pre *= Cg[w]; q[gate][w] = pre; }
            float suf = Cg[7];
            #pragma unroll
            for (int w = 6; w >= 1; --w) suf *= Cg[w];
            q[gate][0] = suf;
        }
        #pragma unroll
        for (int j = 0; j < NQ; ++j) {
            float fg = 1.f / (1.f + __expf(-q[0][j]));
            float ig = 1.f / (1.f + __expf(-q[1][j]));
            float gg = tanhf(q[2][j]);
            float og = 1.f / (1.f + __expf(-q[3][j]));
            c[j] = fg * c[j] + ig * gg;
            h[j] = og * tanhf(c[j]);
        }
        // tag logits + log_softmax
        float l[TAGSET], mx = -1e30f;
        #pragma unroll
        for (int k = 0; k < TAGSET; ++k) {
            float v = sbt[k];
            #pragma unroll
            for (int j = 0; j < HIDDEN; ++j) v = fmaf(h[j], sWt[k * HIDDEN + j], v);
            l[k] = v; mx = fmaxf(mx, v);
        }
        float s = 0.f;
        #pragma unroll
        for (int k = 0; k < TAGSET; ++k) s += __expf(l[k] - mx);
        float lse = mx + __logf(s);
        float* op = out + t * (B_ * TAGSET) + b * TAGSET;
        #pragma unroll
        for (int k = 0; k < TAGSET; ++k) op[k] = l[k] - lse;
    }
}

extern "C" void kernel_launch(void* const* d_in, const int* in_sizes, int n_in,
                              void* d_out, int out_size, void* d_ws, size_t ws_size,
                              hipStream_t stream) {
    (void)in_sizes; (void)n_in; (void)out_size; (void)ws_size;
    const int*   sentence = (const int*)  d_in[0];
    const float* emb      = (const float*)d_in[1];
    const float* Wf  = (const float*)d_in[2];
    const float* bf  = (const float*)d_in[3];
    const float* thf = (const float*)d_in[4];
    const float* Wi  = (const float*)d_in[5];
    const float* bi  = (const float*)d_in[6];
    const float* thi = (const float*)d_in[7];
    const float* Wu  = (const float*)d_in[8];
    const float* bu  = (const float*)d_in[9];
    const float* thu = (const float*)d_in[10];
    const float* Wo  = (const float*)d_in[11];
    const float* bo  = (const float*)d_in[12];
    const float* tho = (const float*)d_in[13];
    const float* Wtag = (const float*)d_in[14];
    const float* btag = (const float*)d_in[15];

    char* ws = (char*)d_ws;
    float*     Xp     = (float*)ws;                                   // 1 MB
    _Float16*  Wx     = (_Float16*)(ws + XP_FLOATS * sizeof(float));  // 4 KB
    float*     biasth = (float*)((char*)Wx + G32 * EMBED * sizeof(_Float16));
    float*     Wh     = biasth + G32;

    qlstm_prep<<<1, 256, 0, stream>>>(Wf, bf, thf, Wi, bi, thi, Wu, bu, thu,
                                      Wo, bo, tho, Wx, biasth, Wh);
    qlstm_xproj<<<128, 128, 0, stream>>>(sentence, emb, Wx, biasth, Xp);
    qlstm_recur<<<4, 32, 0, stream>>>(Xp, Wh, Wtag, btag, (float*)d_out);
}
